// MessagePassingLayer_58059367907338
// MI455X (gfx1250) — matrix operand order repaired and checked by reading, not compiled, not run on hardware
//
#include <hip/hip_runtime.h>
#include <hip/hip_bf16.h>

// MessagePassingLayer for MI455X (gfx1250, wave32, WMMA).
//
// Pipeline (3 kernels, all on `stream`):
//  1) init:  zero agg accumulator (f32) in d_ws, convert W1/W2/w_ih/w_hh -> bf16 in d_ws
//  2) edge:  per 16-edge tile (one wave32): gather [16,160] input in bf16 A-frags,
//            2-layer MLP via v_wmma_f32_16x16x32_bf16 (f32 accum), ReLU in f32,
//            scatter messages with global_atomic_add_f32 into agg (L2-resident).
//  3) gru:   per 16-node tile: gi/gh gate GEMMs via WMMA, f32 sigmoid/tanh/blend.
//
// Workspace layout (bytes): [agg f32 25.6MB][W1b 20KB][W2b 8KB][wihb 24KB][whhb 24KB]

typedef __bf16 bf16_t;
typedef __bf16 v16bf __attribute__((ext_vector_type(16)));
typedef float  v8f   __attribute__((ext_vector_type(8)));

#define NNODES 100000
#define NEDGES 1000000
#define NDIM   64
#define EDIM   32
#define HID    64
#define INDIM  160   // 2*NDIM + EDIM

#define AGG_ELEMS   (NNODES * NDIM)           // 6,400,000 f32
#define W1_ELEMS    (HID * INDIM)             // 10,240
#define W2_ELEMS    (HID * HID)               // 4,096
#define WIH_ELEMS   (3 * NDIM * HID)          // 12,288
#define WHH_ELEMS   (3 * NDIM * NDIM)         // 12,288

__device__ __forceinline__ float sigmoidf_(float x) { return 1.0f / (1.0f + __expf(-x)); }

__device__ __forceinline__ void cvt8(const float* __restrict__ p, bf16_t* o) {
  float4 a = *(const float4*)p;
  float4 b = *(const float4*)(p + 4);
  o[0] = (bf16_t)a.x; o[1] = (bf16_t)a.y; o[2] = (bf16_t)a.z; o[3] = (bf16_t)a.w;
  o[4] = (bf16_t)b.x; o[5] = (bf16_t)b.y; o[6] = (bf16_t)b.z; o[7] = (bf16_t)b.w;
}

// A fragment (16x32 bf16, M x K) for this lane, from a row-major f32 row base.
// ISA layout: lane L holds row L%16; elements 0..7 -> K = laneHi*8 + i,
// elements 8..15 -> K = 16 + laneHi*8 + (i-8).
__device__ __forceinline__ v16bf loadA32(const float* __restrict__ rowbase, int kbase, int laneHi) {
  union { v16bf v; bf16_t e[16]; } u;
  cvt8(rowbase + kbase + laneHi * 8,      u.e);
  cvt8(rowbase + kbase + laneHi * 8 + 16, u.e + 8);
  return u.v;
}

// B fragment (32x16 bf16, K x N): column n = lane%16 maps to weight ROW `ncol`
// (since result = x @ W^T), elements 0..7 -> K = kbase+laneHi*8+i, 8..15 -> K = kbase+16+laneHi*8+(i-8).
__device__ __forceinline__ v16bf loadB32(const bf16_t* __restrict__ W, int ldw,
                                         int ncol, int kbase, int laneHi) {
  union { v16bf v; uint4 q[2]; } u;
  const bf16_t* p = W + (size_t)ncol * ldw + kbase + laneHi * 8;
  u.q[0] = *(const uint4*)(p);
  u.q[1] = *(const uint4*)(p + 16);
  return u.v;
}

__device__ __forceinline__ v8f wmma_bf16(v16bf a, v16bf b, v8f c) {
  return __builtin_amdgcn_wmma_f32_16x16x32_bf16(false, a, false, b, (short)0, c, false, false);
}

// ---------------------------------------------------------------------------
// Kernel 1: init — zero agg, convert weights f32 -> bf16
// ---------------------------------------------------------------------------
__global__ __launch_bounds__(256)
void mp_init_kernel(const float* __restrict__ W1, const float* __restrict__ W2,
                    const float* __restrict__ wih, const float* __restrict__ whh,
                    float* __restrict__ agg,
                    bf16_t* __restrict__ W1b, bf16_t* __restrict__ W2b,
                    bf16_t* __restrict__ wihb, bf16_t* __restrict__ whhb) {
  int i = blockIdx.x * blockDim.x + threadIdx.x;
  if (i < AGG_ELEMS) agg[i] = 0.0f;
  if (i < W1_ELEMS)  W1b[i]  = (bf16_t)W1[i];
  if (i < W2_ELEMS)  W2b[i]  = (bf16_t)W2[i];
  if (i < WIH_ELEMS) wihb[i] = (bf16_t)wih[i];
  if (i < WHH_ELEMS) whhb[i] = (bf16_t)whh[i];
}

// ---------------------------------------------------------------------------
// Kernel 2: edge message MLP + atomic scatter. One wave32 per 16-edge tile.
// ---------------------------------------------------------------------------
__global__ __launch_bounds__(128)
void mp_edge_kernel(const float* __restrict__ nodef,
                    const int*   __restrict__ eidx,   // [2, E] flat, int32
                    const float* __restrict__ edgef,
                    const float* __restrict__ b1,
                    const float* __restrict__ b2,
                    const bf16_t* __restrict__ W1b,
                    const bf16_t* __restrict__ W2b,
                    float* __restrict__ agg,
                    int n_tiles) {
  __shared__ __align__(16) bf16_t hbuf[4][16 * 72]; // per-wave 16x64 hidden, stride 72
  __shared__ int dstsh[4][16];

  const int lane   = threadIdx.x & 31;
  const int wband  = threadIdx.x >> 5;
  const int tile   = blockIdx.x * 4 + wband;
  if (tile >= n_tiles) return;
  const int laneHi = lane >> 4;
  const int laneLo = lane & 15;

  // This lane's input row (edge) for A fragments.
  int eRow = tile * 16 + laneLo;
  bool rowValid = (eRow < NEDGES);
  int eC = rowValid ? eRow : (NEDGES - 1);
  int src = eidx[eC];
  int dst = eidx[NEDGES + eC];
  if (laneHi == 0) dstsh[wband][laneLo] = rowValid ? dst : -1;

  const float* srcP = nodef + (size_t)src * NDIM;
  const float* dstP = nodef + (size_t)dst * NDIM;
  const float* edgP = edgef + (size_t)eC * EDIM;

  // Layer-1 A fragments: x = [node[src](64) | node[dst](64) | edge(32)], K = 160.
  v16bf A[5];
  A[0] = loadA32(srcP, 0,  laneHi);
  A[1] = loadA32(srcP, 32, laneHi);
  A[2] = loadA32(dstP, 0,  laneHi);
  A[3] = loadA32(dstP, 32, laneHi);
  A[4] = loadA32(edgP, 0,  laneHi);

  // Layer 1: h = relu(x @ W1^T + b1), 4 N-tiles of 16 columns.
  #pragma unroll
  for (int nt = 0; nt < 4; ++nt) {
    v8f c = {0.f, 0.f, 0.f, 0.f, 0.f, 0.f, 0.f, 0.f};
    #pragma unroll
    for (int kt = 0; kt < 5; ++kt) {
      v16bf b = loadB32(W1b, INDIM, nt * 16 + laneLo, kt * 32, laneHi);
      c = wmma_bf16(A[kt], b, c);
    }
    float bias = b1[nt * 16 + laneLo];
    #pragma unroll
    for (int r = 0; r < 8; ++r) {
      float h = fmaxf(c[r] + bias, 0.0f);
      // C layout: vgpr r, lanes hold (M = r + 8*laneHi, N = laneLo)
      hbuf[wband][(r + 8 * laneHi) * 72 + nt * 16 + laneLo] = (bf16_t)h;
    }
  }

  // DS ops are in-order per wave; keep the compiler from reordering LDS use.
  asm volatile("" ::: "memory");
  __builtin_amdgcn_wave_barrier();

  // Re-shape hidden tile (C layout -> A layout) via LDS.
  v16bf H[2];
  #pragma unroll
  for (int kt = 0; kt < 2; ++kt) {
    union { v16bf v; uint4 q[2]; } u;
    const bf16_t* base = &hbuf[wband][laneLo * 72 + kt * 32 + laneHi * 8];
    u.q[0] = *(const uint4*)(base);
    u.q[1] = *(const uint4*)(base + 16);
    H[kt] = u.v;
  }

  // Layer 2: messages = h @ W2^T + b2, then atomic scatter by dst.
  #pragma unroll
  for (int nt = 0; nt < 4; ++nt) {
    v8f c = {0.f, 0.f, 0.f, 0.f, 0.f, 0.f, 0.f, 0.f};
    #pragma unroll
    for (int kt = 0; kt < 2; ++kt) {
      v16bf b = loadB32(W2b, HID, nt * 16 + laneLo, kt * 32, laneHi);
      c = wmma_bf16(H[kt], b, c);
    }
    float bias = b2[nt * 16 + laneLo];
    #pragma unroll
    for (int r = 0; r < 8; ++r) {
      int m = r + 8 * laneHi;           // edge row within tile
      int d = dstsh[wband][m];
      if (d >= 0) {
        atomicAdd(&agg[(size_t)d * HID + nt * 16 + laneLo], c[r] + bias);
      }
    }
  }
}

// ---------------------------------------------------------------------------
// Kernel 3: GRU cell. One wave32 per 16-node tile.
// gi = agg @ w_ih^T + b_ih ; gh = h @ w_hh^T + b_hh ; gates [r;z;n]
// ---------------------------------------------------------------------------
__global__ __launch_bounds__(128)
void mp_gru_kernel(const float* __restrict__ nodef,
                   const float* __restrict__ agg,
                   const bf16_t* __restrict__ wihb,
                   const bf16_t* __restrict__ whhb,
                   const float* __restrict__ bih,
                   const float* __restrict__ bhh,
                   float* __restrict__ out,
                   int n_tiles) {
  const int lane   = threadIdx.x & 31;
  const int wband  = threadIdx.x >> 5;
  const int tile   = blockIdx.x * 4 + wband;
  if (tile >= n_tiles) return;
  const int laneHi = lane >> 4;
  const int laneLo = lane & 15;

  int nRow = tile * 16 + laneLo;
  int nC = (nRow < NNODES) ? nRow : (NNODES - 1);
  const float* aggP = agg   + (size_t)nC * NDIM;
  const float* hP   = nodef + (size_t)nC * NDIM;

  v16bf Aa[2], Ah[2];
  Aa[0] = loadA32(aggP, 0, laneHi);  Aa[1] = loadA32(aggP, 32, laneHi);
  Ah[0] = loadA32(hP,   0, laneHi);  Ah[1] = loadA32(hP,   32, laneHi);

  #pragma unroll
  for (int j = 0; j < 4; ++j) {       // 16-column group of the 64-wide gates
    v8f ci[3], ch[3];
    #pragma unroll
    for (int g = 0; g < 3; ++g) {     // r, z, n
      v8f a = {0.f, 0.f, 0.f, 0.f, 0.f, 0.f, 0.f, 0.f};
      v8f h = {0.f, 0.f, 0.f, 0.f, 0.f, 0.f, 0.f, 0.f};
      int ob = g * NDIM + j * 16 + laneLo;  // weight row (output channel)
      #pragma unroll
      for (int kt = 0; kt < 2; ++kt) {
        v16bf bi = loadB32(wihb, HID,  ob, kt * 32, laneHi);
        a = wmma_bf16(Aa[kt], bi, a);
        v16bf bh = loadB32(whhb, NDIM, ob, kt * 32, laneHi);
        h = wmma_bf16(Ah[kt], bh, h);
      }
      ci[g] = a; ch[g] = h;
    }
    int col = j * 16 + laneLo;
    float bir = bih[col],            bhr = bhh[col];
    float biz = bih[NDIM + col],     bhz = bhh[NDIM + col];
    float bin = bih[2 * NDIM + col], bhn = bhh[2 * NDIM + col];
    #pragma unroll
    for (int r = 0; r < 8; ++r) {
      int m = tile * 16 + r + 8 * laneHi;   // node id
      if (m < NNODES) {
        float ir = ci[0][r] + bir, iz = ci[1][r] + biz, in = ci[2][r] + bin;
        float hr = ch[0][r] + bhr, hz = ch[1][r] + bhz, hn = ch[2][r] + bhn;
        float rg = sigmoidf_(ir + hr);
        float zg = sigmoidf_(iz + hz);
        float ng = tanhf(in + rg * hn);
        float hprev = nodef[(size_t)m * NDIM + col];
        out[(size_t)m * NDIM + col] = (1.0f - zg) * ng + zg * hprev;
      }
    }
  }
}

// ---------------------------------------------------------------------------
extern "C" void kernel_launch(void* const* d_in, const int* in_sizes, int n_in,
                              void* d_out, int out_size, void* d_ws, size_t ws_size,
                              hipStream_t stream) {
  const float* nodef = (const float*)d_in[0];
  const int*   eidx  = (const int*)  d_in[1];   // edge_index [2, E] (int32 per harness)
  const float* edgef = (const float*)d_in[2];
  const float* W1    = (const float*)d_in[3];
  const float* b1    = (const float*)d_in[4];
  const float* W2    = (const float*)d_in[5];
  const float* b2    = (const float*)d_in[6];
  const float* wih   = (const float*)d_in[7];
  const float* whh   = (const float*)d_in[8];
  const float* bih   = (const float*)d_in[9];
  const float* bhh   = (const float*)d_in[10];
  float* out = (float*)d_out;

  // Workspace carve-up (all offsets 32B-aligned).
  char* ws = (char*)d_ws;
  float*  agg  = (float*)ws;
  bf16_t* W1b  = (bf16_t*)(ws + (size_t)AGG_ELEMS * 4);
  bf16_t* W2b  = (bf16_t*)((char*)W1b  + (size_t)W1_ELEMS  * 2);
  bf16_t* wihb = (bf16_t*)((char*)W2b  + (size_t)W2_ELEMS  * 2);
  bf16_t* whhb = (bf16_t*)((char*)wihb + (size_t)WIH_ELEMS * 2);

  // 1) zero agg + weight conversion (must run every call: atomics accumulate).
  {
    int threads = 256;
    int blocks = (AGG_ELEMS + threads - 1) / threads;
    mp_init_kernel<<<blocks, threads, 0, stream>>>(W1, W2, wih, whh,
                                                   agg, W1b, W2b, wihb, whhb);
  }
  // 2) edge message MLP + scatter. 4 waves/block, one 16-edge tile per wave.
  {
    int n_tiles = (NEDGES + 15) / 16;          // 62500
    int blocks = (n_tiles + 3) / 4;            // 15625
    mp_edge_kernel<<<blocks, 128, 0, stream>>>(nodef, eidx, edgef, b1, b2,
                                               W1b, W2b, agg, n_tiles);
  }
  // 3) GRU update. One 16-node tile per wave.
  {
    int n_tiles = (NNODES + 15) / 16;          // 6250
    int blocks = (n_tiles + 3) / 4;            // 1563
    mp_gru_kernel<<<blocks, 128, 0, stream>>>(nodef, agg, wihb, whhb,
                                              bih, bhh, out, n_tiles);
  }
}
